// LSTM_1717986919093
// MI455X (gfx1250) — matrix operand order repaired and checked by reading, not compiled, not run on hardware
//
#include <hip/hip_runtime.h>
#include <hip/hip_bf16.h>

typedef __attribute__((ext_vector_type(16))) _Float16 v16h;
typedef __attribute__((ext_vector_type(8)))  _Float16 v8h;
typedef __attribute__((ext_vector_type(8)))  float    v8f;
typedef __attribute__((ext_vector_type(4)))  float    v4f;

static constexpr int LSTM_B  = 256;   // batch
static constexpr int LSTM_T  = 1024;  // timesteps
static constexpr int LSTM_D  = 128;   // input dim
static constexpr int LSTM_H  = 256;   // hidden dim
static constexpr int LSTM_O  = 10;    // output classes
static constexpr int LSTM_G  = 1024;  // 4*H gate columns
static constexpr int LSTM_K  = 384;   // D + H (fused GEMM K)
static constexpr int NBLOCKS = 32;    // persistent grid size

// ---------------------------------------------------------------- helpers
__device__ __forceinline__ v8f splat8(float s) {
  v8f r;
#pragma unroll
  for (int i = 0; i < 8; ++i) r[i] = s;
  return r;
}

__device__ __forceinline__ v16h cat16(v8h lo, v8h hi) {
  return __builtin_shufflevector(lo, hi, 0,1,2,3,4,5,6,7,8,9,10,11,12,13,14,15);
}

__device__ __forceinline__ v8h cvt8(v4f a, v4f b) {
  v8h r;
  r[0] = (_Float16)a[0]; r[1] = (_Float16)a[1];
  r[2] = (_Float16)a[2]; r[3] = (_Float16)a[3];
  r[4] = (_Float16)b[0]; r[5] = (_Float16)b[1];
  r[6] = (_Float16)b[2]; r[7] = (_Float16)b[3];
  return r;
}

// Native V_TANH_F32 on gfx1250 (single TRANS op, co-executes with VALU);
// guarded fallback uses tanh(x) = 1 - 2/(exp(2x)+1) -> one v_exp_f32.
__device__ __forceinline__ float tanh_f(float x) {
#if __has_builtin(__builtin_amdgcn_tanhf)
  return __builtin_amdgcn_tanhf(x);
#else
  return 1.0f - 2.0f / (__expf(2.0f * x) + 1.0f);
#endif
}

__device__ __forceinline__ float sigmoid_f(float x) {
  return 1.0f / (1.0f + __expf(-x));
}

// ---------------------------------------------------------------- prep kernels
struct GateWeights {
  const float* wx[4];  // [D,H] each, gate order g,i,f,o
  const float* bx[4];  // [H]
  const float* wh[4];  // [H,H]
  const float* bh[4];  // [H]
};

// Wt[n][k] = concat(Wx;Wh)[k][n] as f16  (N-major so WMMA B-frags are contiguous)
// bias[n]  = bx[n] + bh[n]
__global__ void lstm_prep_weights(GateWeights gw, _Float16* __restrict__ Wt,
                                  float* __restrict__ bias) {
  int idx = blockIdx.x * blockDim.x + threadIdx.x;
  if (idx >= LSTM_G * LSTM_K) return;
  int n = idx / LSTM_K, k = idx % LSTM_K;
  int gi = n >> 8, hc = n & 255;
  float v = (k < LSTM_D) ? gw.wx[gi][k * LSTM_H + hc]
                         : gw.wh[gi][(k - LSTM_D) * LSTM_H + hc];
  Wt[(size_t)n * LSTM_K + k] = (_Float16)v;
  if (k == 0) bias[n] = gw.bx[gi][hc] + gw.bh[gi][hc];
}

__global__ void lstm_prep_state(_Float16* __restrict__ h0,
                                _Float16* __restrict__ h1,
                                unsigned* __restrict__ counter) {
  int idx = blockIdx.x * blockDim.x + threadIdx.x;
  if (idx < LSTM_B * LSTM_H) {
    h0[idx] = (_Float16)0.0f;
    h1[idx] = (_Float16)0.0f;
  }
  if (idx == 0) *counter = 0u;
}

// ---------------------------------------------------------------- persistent LSTM
// Grid: 32 WGs x 256 threads. blockIdx = rg(0..1)*16 + cg(0..15).
//   WG covers rows [128*rg, 128*rg+127] x hidden cols [16*cg, 16*cg+15].
//   Wave wv (0..7) owns rows 128*rg + 16*wv .. +15; computes all 4 gate tiles,
//   keeps its 16x16 cell-state tile in registers for the entire T loop.
//   (Compiler hoists all 48 loop-invariant Wt B-fragments into VGPRs.)
__global__ __launch_bounds__(256, 1) void lstm_persistent(
    const float* __restrict__ x,       // [B,T,D] f32
    const _Float16* __restrict__ Wt,   // [1024][384] f16 (N-major)
    const float* __restrict__ bias,    // [1024] f32 (bx+bh)
    _Float16* __restrict__ hbuf0,      // [B,H] f16 ping
    _Float16* __restrict__ hbuf1,      // [B,H] f16 pong
    unsigned* __restrict__ counter) {
  const int tid     = threadIdx.x;
  const int lane    = tid & 31;
  const int wv      = tid >> 5;
  const int half_id = lane >> 4;   // lane group (0: lanes 0-15, 1: lanes 16-31)
  const int l15     = lane & 15;
  const int rg      = blockIdx.x >> 4;
  const int cg      = blockIdx.x & 15;
  const int rowbase = rg * 128 + wv * 16;
  const int arow    = rowbase + l15;        // A-fragment row (batch index)
  const int ncol    = cg * 16 + l15;        // B/C/D column (hidden index)

  const float* xrow = x + (size_t)arow * LSTM_T * LSTM_D;

  const _Float16* wrow[4];
  float bval[4];
#pragma unroll
  for (int gi = 0; gi < 4; ++gi) {
    wrow[gi] = Wt + (size_t)(gi * LSTM_H + ncol) * LSTM_K;
    bval[gi] = bias[gi * LSTM_H + ncol];
  }

  v8f cstate = {};  // cell state tile, lives in registers for all T steps

  for (int t = 0; t < LSTM_T; ++t) {
    const _Float16* hprev = (t & 1) ? hbuf1 : hbuf0;
    _Float16*       hnext = (t & 1) ? hbuf0 : hbuf1;

    v8f acc[4];
#pragma unroll
    for (int gi = 0; gi < 4; ++gi) acc[gi] = splat8(bval[gi]);

    const float* xp = xrow + (size_t)t * LSTM_D;
    if (t + 1 < LSTM_T)  // pull next step's x row toward L2 (global_prefetch_b8)
      __builtin_prefetch(xp + LSTM_D + 8 * half_id, 0, 0);

    // ---- K-steps 0..3: x contribution (f32 -> f16 in flight) ----
#pragma unroll
    for (int ks = 0; ks < 4; ++ks) {
      const int kb = ks * 32;
      // A frag (16x32 f16): elems 0-7 -> K = kb + 8*half + e ; 8-15 -> +16
      const float* p0 = xp + kb + 8 * half_id;
      const float* p1 = xp + kb + 16 + 8 * half_id;
      v4f f0 = *(const v4f*)(p0);
      v4f f1 = *(const v4f*)(p0 + 4);
      v4f f2 = *(const v4f*)(p1);
      v4f f3 = *(const v4f*)(p1 + 4);
      v16h a = cat16(cvt8(f0, f1), cvt8(f2, f3));
#pragma unroll
      for (int gi = 0; gi < 4; ++gi) {
        // B frag (32x16 f16): lane = N, elems 0-7 -> K = kb + 8*half + e ; 8-15 -> +16
        const _Float16* wp = wrow[gi] + kb + 8 * half_id;
        v16h bfrag = cat16(*(const v8h*)(wp), *(const v8h*)(wp + 16));
        acc[gi] = __builtin_amdgcn_wmma_f32_16x16x32_f16(
            false, a, false, bfrag, (short)0, acc[gi], false, false);
      }
    }

    // ---- K-steps 4..11: h contribution (f16 direct) ----
    const _Float16* hrow = hprev + (size_t)arow * LSTM_H;
#pragma unroll
    for (int ks = 4; ks < 12; ++ks) {
      const int kb = ks * 32;
      const int kh = kb - LSTM_D;
      v8h a0 = *(const v8h*)(hrow + kh + 8 * half_id);
      v8h a1 = *(const v8h*)(hrow + kh + 16 + 8 * half_id);
      v16h a = cat16(a0, a1);
#pragma unroll
      for (int gi = 0; gi < 4; ++gi) {
        const _Float16* wp = wrow[gi] + kb + 8 * half_id;
        v16h bfrag = cat16(*(const v8h*)(wp), *(const v8h*)(wp + 16));
        acc[gi] = __builtin_amdgcn_wmma_f32_16x16x32_f16(
            false, a, false, bfrag, (short)0, acc[gi], false, false);
      }
    }

    // ---- elementwise cell update; D layout: elem v -> row v + 8*half, col = ncol ----
    _Float16* hcolp = hnext + ncol;
#pragma unroll
    for (int v = 0; v < 8; ++v) {
      float g  = tanh_f(acc[0][v]);
      float ig = sigmoid_f(acc[1][v]);
      float fg = sigmoid_f(acc[2][v]);
      float og = sigmoid_f(acc[3][v]);
      float cn = g * ig + cstate[v] * fg;
      cstate[v] = cn;
      float hv = tanh_f(cn) * og;
      int row = rowbase + v + 8 * half_id;
      hcolp[(size_t)row * LSTM_H] = (_Float16)hv;
    }

    // ---- device-wide barrier: everyone's h written before next step reads ----
    __threadfence();
    __syncthreads();
    if (tid == 0) {
      __hip_atomic_fetch_add(counter, 1u, __ATOMIC_ACQ_REL, __HIP_MEMORY_SCOPE_AGENT);
      const unsigned target = (unsigned)(t + 1) * NBLOCKS;
      while (__hip_atomic_load(counter, __ATOMIC_ACQUIRE, __HIP_MEMORY_SCOPE_AGENT) < target)
        __builtin_amdgcn_s_sleep(2);
    }
    __syncthreads();
    __threadfence();
  }
}

// ---------------------------------------------------------------- epilogue: p = h_T @ Wph + bph, softmax
__global__ void lstm_final_proj(const _Float16* __restrict__ hT,  // [B,H] f16 (hbuf0)
                                const float* __restrict__ Wph,    // [H,O]
                                const float* __restrict__ bph,    // [O]
                                float* __restrict__ out) {        // [B,O]
  int b = blockIdx.x * blockDim.x + threadIdx.x;
  if (b >= LSTM_B) return;
  float acc[LSTM_O];
#pragma unroll
  for (int j = 0; j < LSTM_O; ++j) acc[j] = bph[j];
  const _Float16* hr = hT + (size_t)b * LSTM_H;
  for (int k = 0; k < LSTM_H; ++k) {
    float hv = (float)hr[k];
    const float* wr = Wph + (size_t)k * LSTM_O;
#pragma unroll
    for (int j = 0; j < LSTM_O; ++j) acc[j] += hv * wr[j];
  }
  float mx = acc[0];
#pragma unroll
  for (int j = 1; j < LSTM_O; ++j) mx = fmaxf(mx, acc[j]);
  float s = 0.0f;
#pragma unroll
  for (int j = 0; j < LSTM_O; ++j) { acc[j] = __expf(acc[j] - mx); s += acc[j]; }
  float inv = 1.0f / s;
#pragma unroll
  for (int j = 0; j < LSTM_O; ++j) out[(size_t)b * LSTM_O + j] = acc[j] * inv;
}

// ---------------------------------------------------------------- launch
extern "C" void kernel_launch(void* const* d_in, const int* in_sizes, int n_in,
                              void* d_out, int out_size, void* d_ws, size_t ws_size,
                              hipStream_t stream) {
  (void)in_sizes; (void)n_in; (void)out_size; (void)ws_size;
  // dict order: x, then per gate (g,i,f,o): W?x, b?x, W?h, b?h, then Wph, bph
  const float* x = (const float*)d_in[0];
  GateWeights gw;
  for (int gi = 0; gi < 4; ++gi) {
    int base = 1 + gi * 4;
    gw.wx[gi] = (const float*)d_in[base + 0];
    gw.bx[gi] = (const float*)d_in[base + 1];
    gw.wh[gi] = (const float*)d_in[base + 2];
    gw.bh[gi] = (const float*)d_in[base + 3];
  }
  const float* Wph = (const float*)d_in[17];
  const float* bph = (const float*)d_in[18];

  // workspace layout (~1.05 MB)
  char* ws = (char*)d_ws;
  _Float16* Wt   = (_Float16*)(ws);                                  // 1024*384*2 = 786432
  float*    bias = (float*)(ws + 786432);                            // 4096
  _Float16* h0   = (_Float16*)(ws + 786432 + 4096);                  // 131072
  _Float16* h1   = (_Float16*)(ws + 786432 + 4096 + 131072);         // 131072
  unsigned* cnt  = (unsigned*)(ws + 786432 + 4096 + 2 * 131072);     // 4

  lstm_prep_weights<<<(LSTM_G * LSTM_K + 255) / 256, 256, 0, stream>>>(gw, Wt, bias);
  lstm_prep_state<<<(LSTM_B * LSTM_H + 255) / 256, 256, 0, stream>>>(h0, h1, cnt);
  lstm_persistent<<<NBLOCKS, 256, 0, stream>>>(x, Wt, bias, h0, h1, cnt);
  lstm_final_proj<<<1, 256, 0, stream>>>(h0, Wph, bph, (float*)d_out);
}
